// GENTAG_37967510896760
// MI455X (gfx1250) — matrix operand order repaired and checked by reading, not compiled, not run on hardware
//
#include <hip/hip_runtime.h>
#include <hip/hip_bf16.h>
#include <math.h>

// ---------------------------------------------------------------------------
// TAGConv x2 GNN for MI455X (gfx1250, wave32).
//  - Edge propagation: wave-per-edge, float4 gather + global_atomic_add_f32
//    scatter (both feature arrays L2-resident: 25.6MB << 192MB L2).
//  - Dense math: v_wmma_f32_16x16x32_bf16. Weights pre-packed once into the
//    WMMA B VGPR layout (bf16, contiguous 32B per lane). Each wave computes a
//    16x64 output strip: A fragment loaded once per K-step, reused by 4 WMMAs.
//  - bias / frozen-BN / ReLU fused into GEMM epilogues.
// ---------------------------------------------------------------------------

typedef __attribute__((ext_vector_type(16))) __bf16 v16bf;
typedef __attribute__((ext_vector_type(8)))  float  v8f;

#define GEPS      1e-7f
#define BN_SCALE  0.99999500003749969f   // 1/sqrt(1 + 1e-5)
#define TN        4                      // 16-col n-tiles per wave (64 cols)

__device__ __forceinline__ void atomic_add_f32(float* p, float v) {
    __hip_atomic_fetch_add(p, v, __ATOMIC_RELAXED, __HIP_MEMORY_SCOPE_AGENT);
}

// ---------------------------- utility kernels ------------------------------

__global__ __launch_bounds__(256) void k_zero(float* __restrict__ p, long n) {
    long i = (long)blockIdx.x * blockDim.x + threadIdx.x;
    if (i < n) p[i] = 0.0f;
}

__global__ __launch_bounds__(256) void k_degree(const int* __restrict__ cols,
                                                float* __restrict__ deg, int E) {
    int e = blockIdx.x * blockDim.x + threadIdx.x;
    if (e < E) atomic_add_f32(&deg[cols[e]], 1.0f);
}

__global__ __launch_bounds__(256) void k_dis(float* __restrict__ deg, int n) {
    int i = blockIdx.x * blockDim.x + threadIdx.x;
    if (i < n) {
        float d = deg[i];
        deg[i] = (d > 0.0f) ? rsqrtf(fmaxf(d, 1.0f)) : 0.0f;
    }
}

__global__ __launch_bounds__(256) void k_norm(const int* __restrict__ rows,
                                              const int* __restrict__ cols,
                                              const float* __restrict__ dis,
                                              float* __restrict__ nrm, int E) {
    int e = blockIdx.x * blockDim.x + threadIdx.x;
    if (e < E) nrm[e] = dis[rows[e]] * dis[cols[e]];
}

// -------------------- weight pre-pack into WMMA B layout --------------------
// W[K x Nc] f32 row-major  ->  Bp[(kb*ntn + nt)*32 + lane][16] bf16, where
// lane half 0 holds K = kb*32 + 0..15, half 1 holds K = kb*32 + 16..31,
// column = nt*16 + (lane & 15). One 32B store per thread.

__global__ __launch_bounds__(256) void k_pack_w(const float* __restrict__ W,
                                                __bf16* __restrict__ Bp,
                                                int K, int Nc) {
    const int total = (K >> 5) * (Nc >> 4) * 32;
    const int idx = blockIdx.x * blockDim.x + threadIdx.x;
    if (idx >= total) return;
    const int lane = idx & 31;
    const int tile = idx >> 5;           // kb * ntn + nt
    const int ntn  = Nc >> 4;
    const int kb   = tile / ntn;
    const int nt   = tile % ntn;
    const int half = lane >> 4;
    const int l16  = lane & 15;
    const float* src = W + (size_t)(kb * 32 + half * 16) * Nc + nt * 16 + l16;
    __bf16* dst = Bp + (size_t)idx * 16;
#pragma unroll
    for (int e = 0; e < 16; ++e) dst[e] = (__bf16)src[(size_t)e * Nc];
}

// --------------------- edge propagation (128 channels) ---------------------
// One wave32 per edge: 32 lanes x float4 = 128 channels.

__global__ __launch_bounds__(256) void k_hop128(const int* __restrict__ rows,
                                                const int* __restrict__ cols,
                                                const float* __restrict__ nrm,
                                                const float* __restrict__ hsrc,
                                                float* __restrict__ hdst, int E) {
    const int wid = blockIdx.x * (blockDim.x >> 5) + (threadIdx.x >> 5);
    if (wid >= E) return;
    const int lane = threadIdx.x & 31;
    const int r = rows[wid];
    const int c = cols[wid];
    const float w = nrm[wid];
    const float4 v = *(const float4*)(hsrc + (size_t)r * 128 + lane * 4);
    float* dp = hdst + (size_t)c * 128 + lane * 4;
    atomic_add_f32(dp + 0, fmaxf(w * v.x, 0.0f) + GEPS);
    atomic_add_f32(dp + 1, fmaxf(w * v.y, 0.0f) + GEPS);
    atomic_add_f32(dp + 2, fmaxf(w * v.z, 0.0f) + GEPS);
    atomic_add_f32(dp + 3, fmaxf(w * v.w, 0.0f) + GEPS);
}

// ------------------------------ WMMA GEMM ----------------------------------
// Out[M x Nc] (+)= A[M x K] * W[K x Nc].  A: f32 row-major (converted to bf16
// in-register). W: pre-packed bf16 (k_pack_w). One wave per 16x64 strip
// (TN=4 accumulators). Requires M%16==0, K%32==0, Nc%64==0 (all true here).

__global__ __launch_bounds__(256) void k_wmma_gemm(
    const float* __restrict__ A, const __bf16* __restrict__ Bp,
    float* __restrict__ Out, int M, int K, int Nc, int accumulate,
    const float* __restrict__ colbias,
    const float* __restrict__ bng, const float* __restrict__ bnb,
    int do_relu)
{
    const int lane    = threadIdx.x & 31;
    const int wave    = blockIdx.x * (blockDim.x >> 5) + (threadIdx.x >> 5);
    const int ntn     = Nc >> 4;            // 16-wide n-tiles
    const int nstrips = ntn / TN;           // 64-wide strips
    const int strips  = (M >> 4) * nstrips;
    if (wave >= strips) return;

    const int m0    = (wave / nstrips) << 4;
    const int nt0   = (wave % nstrips) * TN;   // first n-tile index
    const int half  = lane >> 4;               // 0: lanes 0-15, 1: lanes 16-31
    const int l16   = lane & 15;
    const int arow  = m0 + l16;

    v8f c[TN] = {};
    if (accumulate) {
#pragma unroll
        for (int t = 0; t < TN; ++t) {
            const int bcol = nt0 * 16 + t * 16 + l16;
#pragma unroll
            for (int v = 0; v < 8; ++v)
                c[t][v] = Out[(size_t)(m0 + half * 8 + v) * Nc + bcol];
        }
    }

    const float* arowp = A + (size_t)arow * K;
    for (int k0 = 0; k0 < K; k0 += 32) {
        // A 16x32 bf16, contiguous k order as in Bp: half 0 -> K {0..15}, half 1 -> {16..31}
        const float* ap = arowp + k0 + half * 16;
        const float4 a0 = *(const float4*)(ap + 0);
        const float4 a1 = *(const float4*)(ap + 4);
        const float4 a2 = *(const float4*)(ap + 8);
        const float4 a3 = *(const float4*)(ap + 12);
        v16bf av;
        av[0]  = (__bf16)a0.x; av[1]  = (__bf16)a0.y; av[2]  = (__bf16)a0.z; av[3]  = (__bf16)a0.w;
        av[4]  = (__bf16)a1.x; av[5]  = (__bf16)a1.y; av[6]  = (__bf16)a1.z; av[7]  = (__bf16)a1.w;
        av[8]  = (__bf16)a2.x; av[9]  = (__bf16)a2.y; av[10] = (__bf16)a2.z; av[11] = (__bf16)a2.w;
        av[12] = (__bf16)a3.x; av[13] = (__bf16)a3.y; av[14] = (__bf16)a3.z; av[15] = (__bf16)a3.w;

        const __bf16* bbase = Bp + ((size_t)((k0 >> 5) * ntn + nt0) * 32 + lane) * 16;
#pragma unroll
        for (int t = 0; t < TN; ++t) {
            const v16bf bv = *(const v16bf*)(bbase + (size_t)t * 32 * 16);
            c[t] = __builtin_amdgcn_wmma_f32_16x16x32_bf16(
                       /*neg_a=*/false, av, /*neg_b=*/false, bv,
                       /*c_mod=*/(short)0, c[t], /*reuse_a=*/false, /*reuse_b=*/false);
        }
    }

#pragma unroll
    for (int t = 0; t < TN; ++t) {
        const int bcol = nt0 * 16 + t * 16 + l16;
        const float cb = colbias ? colbias[bcol] : 0.0f;
        float g = 1.0f, bshift = 0.0f;
        const bool bn = (bng != nullptr);
        if (bn) { g = bng[bcol] * BN_SCALE; bshift = bnb[bcol]; }
#pragma unroll
        for (int v = 0; v < 8; ++v) {
            float x = c[t][v] + cb;
            if (bn) x = g * x + bshift;
            if (do_relu) x = fmaxf(x, 0.0f);
            Out[(size_t)(m0 + half * 8 + v) * Nc + bcol] = x;
        }
    }
}

// ------------------------------ host side ----------------------------------

static inline void launch_gemm(const float* A, const __bf16* Bp, float* Out,
                               int M, int K, int Nc, int accumulate,
                               const float* cb, const float* g, const float* b,
                               int relu, hipStream_t s) {
    int strips = (M / 16) * (Nc / (TN * 16));
    int blocks = (strips + 7) / 8;             // 8 waves / 256-thread block
    k_wmma_gemm<<<blocks, 256, 0, s>>>(A, Bp, Out, M, K, Nc, accumulate, cb, g, b, relu);
}

static inline void launch_pack(const float* W, __bf16* Bp, int K, int Nc,
                               hipStream_t s) {
    int total = (K / 32) * (Nc / 16) * 32;
    k_pack_w<<<(total + 255) / 256, 256, 0, s>>>(W, Bp, K, Nc);
}

static inline void launch_zero(float* p, long n, hipStream_t s) {
    k_zero<<<(int)((n + 255) / 256), 256, 0, s>>>(p, n);
}

extern "C" void kernel_launch(void* const* d_in, const int* in_sizes, int n_in,
                              void* d_out, int out_size, void* d_ws, size_t ws_size,
                              hipStream_t stream) {
    const float* x     = (const float*)d_in[0];
    const int*   ei    = (const int*)  d_in[1];
    const float* lins1 = (const float*)d_in[2];
    const float* bias1 = (const float*)d_in[3];
    const float* m1w1  = (const float*)d_in[4];
    const float* m1b1  = (const float*)d_in[5];
    const float* bn1g  = (const float*)d_in[6];
    const float* bn1b  = (const float*)d_in[7];
    const float* m1w2  = (const float*)d_in[8];
    const float* m1b2  = (const float*)d_in[9];
    const float* lins2 = (const float*)d_in[10];
    const float* bias2 = (const float*)d_in[11];
    const float* m2w1  = (const float*)d_in[12];
    const float* m2b1  = (const float*)d_in[13];
    const float* bn2g  = (const float*)d_in[14];
    const float* bn2b  = (const float*)d_in[15];
    const float* m2w2  = (const float*)d_in[16];
    const float* m2b2  = (const float*)d_in[17];

    const int F = 128, H = 128, C = 64, KH = 6;
    const int N = in_sizes[0] / F;        // 50000 (multiple of 16)
    const int E = in_sizes[1] / 2;        // 800000
    const int* rows = ei;
    const int* cols = ei + E;

    // workspace carve-out
    char*  ws  = (char*)d_ws;
    size_t off = 0;
    auto alloc = [&](size_t bytes) -> void* {
        void* p = (void*)(ws + off);
        off = (off + bytes + 255) & ~(size_t)255;
        return p;
    };
    float* deg  = (float*)alloc((size_t)N * 4);        // degree -> dis (in place)
    float* nrm  = (float*)alloc((size_t)E * 4);        // per-edge norm
    float* bufA = (float*)alloc((size_t)N * 128 * 4);  // hop ping
    float* bufB = (float*)alloc((size_t)N * 128 * 4);  // hop pong
    float* acc  = (float*)alloc((size_t)N * 128 * 4);  // TAG accumulator
    float* bufZ = (float*)alloc((size_t)N * 256 * 4);  // MLP hidden
    float* h1   = (float*)alloc((size_t)N * 128 * 4);  // layer-1 output
    // packed bf16 weights (WMMA B layout)
    __bf16* pl1  = (__bf16*)alloc((size_t)(KH + 1) * F * H * 2);
    __bf16* pw1a = (__bf16*)alloc((size_t)H * 2 * H * 2);
    __bf16* pw1b = (__bf16*)alloc((size_t)2 * H * H * 2);
    __bf16* pl2  = (__bf16*)alloc((size_t)(KH + 1) * H * C * 2);
    __bf16* pw2a = (__bf16*)alloc((size_t)C * 2 * C * 2);
    __bf16* pw2b = (__bf16*)alloc((size_t)2 * C * C * 2);
    (void)ws_size; (void)n_in; (void)out_size;

    const int eb = (E + 255) / 256;
    const int nb = (N + 255) / 256;
    const int hopBlocks = (E + 7) / 8;           // 8 waves (edges) per block

    // pack all weights into WMMA B layout (tiny, once per launch)
    for (int k = 0; k <= KH; ++k) {
        launch_pack(lins1 + (size_t)k * F * H, pl1 + (size_t)k * F * H, F, H, stream);
        launch_pack(lins2 + (size_t)k * H * C, pl2 + (size_t)k * H * C, H, C, stream);
    }
    launch_pack(m1w1, pw1a, H, 2 * H, stream);
    launch_pack(m1w2, pw1b, 2 * H, H, stream);
    launch_pack(m2w1, pw2a, C, 2 * C, stream);
    launch_pack(m2w2, pw2b, 2 * C, C, stream);

    // gcn_norm: deg -> dis -> per-edge norm
    launch_zero(deg, N, stream);
    k_degree<<<eb, 256, 0, stream>>>(cols, deg, E);
    k_dis<<<nb, 256, 0, stream>>>(deg, N);
    k_norm<<<eb, 256, 0, stream>>>(rows, cols, deg, nrm, E);

    // ---------------- layer 1: TAGConv(F=128 -> H=128) + MLP ----------------
    launch_gemm(x, pl1, acc, N, F, H, 0, nullptr, nullptr, nullptr, 0, stream);
    {
        const float* hcur = x;
        float* hnext = bufA;
        for (int k = 1; k <= KH; ++k) {
            launch_zero(hnext, (long)N * 128, stream);
            k_hop128<<<hopBlocks, 256, 0, stream>>>(rows, cols, nrm, hcur, hnext, E);
            launch_gemm(hnext, pl1 + (size_t)k * F * H, acc, N, F, H, 1,
                        (k == KH) ? bias1 : nullptr, nullptr, nullptr, 0, stream);
            hcur  = hnext;
            hnext = (hnext == bufA) ? bufB : bufA;
        }
    }
    // MLP1: Linear -> BN -> ReLU -> Linear, then inter-layer ReLU (fused)
    launch_gemm(acc, pw1a, bufZ, N, H, 2 * H, 0, m1b1, bn1g, bn1b, 1, stream);
    launch_gemm(bufZ, pw1b, h1, N, 2 * H, H, 0, m1b2, nullptr, nullptr, 1, stream);

    // ---------------- layer 2: TAGConv(H=128 -> C=64) + MLP -----------------
    launch_gemm(h1, pl2, acc, N, H, C, 0, nullptr, nullptr, nullptr, 0, stream);
    {
        const float* hcur = h1;
        float* hnext = bufA;
        for (int k = 1; k <= KH; ++k) {
            launch_zero(hnext, (long)N * 128, stream);
            k_hop128<<<hopBlocks, 256, 0, stream>>>(rows, cols, nrm, hcur, hnext, E);
            launch_gemm(hnext, pl2 + (size_t)k * H * C, acc, N, H, C, 1,
                        (k == KH) ? bias2 : nullptr, nullptr, nullptr, 0, stream);
            hcur  = hnext;
            hnext = (hnext == bufA) ? bufB : bufA;
        }
    }
    // MLP2: Linear -> BN -> ReLU -> Linear (no final ReLU), straight to d_out
    launch_gemm(acc, pw2a, bufZ, N, C, 2 * C, 0, m2b1, bn2g, bn2b, 1, stream);
    launch_gemm(bufZ, pw2b, (float*)d_out, N, 2 * C, C, 0, m2b2, nullptr, nullptr, 0, stream);
}
